// CoupledHeteroGNN_40492951666821
// MI455X (gfx1250) — matrix operand order repaired and checked by reading, not compiled, not run on hardware
//
#include <hip/hip_runtime.h>
#include <hip/hip_bf16.h>

// ---------------------------------------------------------------------------
// CoupledHeteroGNN for gfx1250 (MI455X).
// Dense per-node GEMMs (N x 128 @ 128 x 128) run on v_wmma_f32_16x16x32_bf16,
// with the weight tile staged to LDS via CDNA5 async global->LDS copies
// (global_load_async_to_lds_b128 + s_wait_asynccnt). sched_group_barrier
// pins a 16x ds_load -> 8x wmma issue pattern per k-step so LDS latency
// overlaps the matrix pipe instead of gating every WMMA.
// x_dst @ W_dst is algebraically folded into a 128x4 GEMV (only its a_d
// projection is ever used), halving the big-GEMM work.
// Segment softmax over edges uses order-preserving u32 encoding + atomicMax,
// then atomicAdd for the normalizer and the weighted scatter.
// ---------------------------------------------------------------------------

typedef unsigned short ushort_t;
typedef unsigned int uint_t;

typedef __attribute__((ext_vector_type(16))) __bf16 v16bf;
typedef __attribute__((ext_vector_type(8)))  float  v8f;

#define HID 128
#define NHEAD 4
#define CH 32

#define CDIV(a, b) (((a) + (b) - 1) / (b))

// Use CDNA5 async global->LDS copy path for weight staging.
#define USE_ASYNC_LDS 1

#if defined(__has_builtin)
#if __has_builtin(__builtin_amdgcn_sched_group_barrier)
#define SCHED_GROUP(mask, size, id) __builtin_amdgcn_sched_group_barrier(mask, size, id)
#endif
#endif
#ifndef SCHED_GROUP
#define SCHED_GROUP(mask, size, id)
#endif

// ---------------- small device helpers ----------------

__device__ __forceinline__ ushort_t f2bf(float x) {
  unsigned u = __float_as_uint(x);
  unsigned r = (u + 0x7FFFu + ((u >> 16) & 1u)) >> 16;  // round-nearest-even
  return (ushort_t)r;
}

// order-preserving float <-> u32 (for atomicMax-based segment max)
__device__ __forceinline__ uint_t enc_f32(float f) {
  uint_t u = __float_as_uint(f);
  return (u & 0x80000000u) ? ~u : (u | 0x80000000u);
}
__device__ __forceinline__ float dec_f32(uint_t e) {
  if (e == 0u) return 0.0f;  // "no incoming edge": reference maps -inf -> 0
  uint_t u = (e & 0x80000000u) ? (e ^ 0x80000000u) : ~e;
  return __uint_as_float(u);
}

// ---------------- kernels ----------------

// y[n,128] = x[n,kin] @ W[kin,128] + b
__global__ void k_inproj(const float* __restrict__ x, const float* __restrict__ W,
                         const float* __restrict__ b, float* __restrict__ y,
                         int n, int kin) {
  long long idx = (long long)blockIdx.x * blockDim.x + threadIdx.x;
  if (idx >= (long long)n * HID) return;
  int row = (int)(idx >> 7), f = (int)(idx & 127);
  float acc = b[f];
  for (int k = 0; k < kin; ++k) acc += x[(long long)row * kin + k] * W[k * HID + f];
  y[idx] = acc;
}

// bf16 conversion
__global__ void k_cvt_bf16(const float* __restrict__ x, ushort_t* __restrict__ y,
                           long long count) {
  long long i = (long long)blockIdx.x * blockDim.x + threadIdx.x;
  if (i < count) y[i] = f2bf(x[i]);
}

// Wt[n,k] = bf16(W[k,n])   (128 x 128)
__global__ void k_wt(const float* __restrict__ W, ushort_t* __restrict__ Wt) {
  int i = blockIdx.x * blockDim.x + threadIdx.x;
  if (i >= HID * HID) return;
  int n = i >> 7, k = i & 127;
  Wt[n * HID + k] = f2bf(W[k * HID + n]);
}

// ------------- WMMA GEMM: Y[rows,128] = Xb[rows,128] @ Wt^T ----------------
// Xb row-major bf16; Wt is [n][k] bf16 (pre-transposed weights), staged into
// LDS once per workgroup (32 KB of the WGP's 320 KB).
union AFrag { v16bf v; uint4 q[2]; };
union CFrag { v8f v; float f[8]; };

__global__ __launch_bounds__(256) void k_gemm_wmma(
    const ushort_t* __restrict__ Xb, const ushort_t* __restrict__ Wt,
    float* __restrict__ Y, int rows) {
  extern __shared__ ushort_t smem[];  // 128*128 bf16 = 32 KB; LDS byte base 0

  // ---- stage Wt into LDS (whole workgroup cooperates, before any exit) ----
#if USE_ASYNC_LDS
#pragma unroll
  for (int j = 0; j < 8; ++j) {
    int c = threadIdx.x + 256 * j;                       // 16B chunk index
    unsigned lds_off = (unsigned)(c * 16);               // LDS byte address
    unsigned long long ga = (unsigned long long)(Wt + c * 8);
    asm volatile("global_load_async_to_lds_b128 %0, %1, off"
                 :: "v"(lds_off), "v"(ga) : "memory");
  }
  asm volatile("s_wait_asynccnt 0" ::: "memory");
#else
#pragma unroll
  for (int j = 0; j < 8; ++j) {
    int c = threadIdx.x + 256 * j;
    *(uint4*)&smem[c * 8] = *(const uint4*)(Wt + c * 8);
  }
#endif
  __syncthreads();

  const int lane = threadIdx.x & 31;
  const int wave = threadIdx.x >> 5;
  const int strip = blockIdx.x * 8 + wave;       // one 16-row strip per wave
  if (strip * 16 >= rows) return;                 // uniform per wave: EXEC stays full

  const int r    = lane & 15;
  const int hsel = lane >> 4;                     // half-wave select
  const long long rowBase = (long long)(strip * 16 + r) * HID;

  // A fragments for the 4 k-steps (ISA 16-bit A 16x32 layout:
  // lane l holds row l&15; halves e: k = (e>>3)*16 + (l>>4)*8 + (e&7))
  AFrag a[4];
#pragma unroll
  for (int s = 0; s < 4; ++s) {
    const ushort_t* ap = Xb + rowBase + s * 32 + hsel * 8;
    a[s].q[0] = *(const uint4*)(ap);        // e = 0..7
    a[s].q[1] = *(const uint4*)(ap + 16);   // e = 8..15
  }

  const int n0 = lane & 15;
  CFrag acc[8];
#pragma unroll
  for (int nt = 0; nt < 8; ++nt)
#pragma unroll
    for (int i = 0; i < 8; ++i) acc[nt].f[i] = 0.0f;

#pragma unroll
  for (int s = 0; s < 4; ++s) {
    // Batch-load all 8 n-tile B fragments of this k-step from LDS, then run
    // the 8 WMMAs.
    AFrag b[8];
#pragma unroll
    for (int nt = 0; nt < 8; ++nt) {
      // B 32x16 layout: lane l -> n = l&15, halves e: k = k0 + (e>>3)*16 + (l>>4)*8 + (e&7)
      const ushort_t* bp = smem + (nt * 16 + n0) * HID + s * 32 + hsel * 8;
      b[nt].q[0] = *(const uint4*)(bp);       // ds_load_b128
      b[nt].q[1] = *(const uint4*)(bp + 16);
    }
#pragma unroll
    for (int nt = 0; nt < 8; ++nt) {
      acc[nt].v = __builtin_amdgcn_wmma_f32_16x16x32_bf16(
          false, a[s].v, false, b[nt].v, (short)0, acc[nt].v, false, false);
    }
    // Pin the issue pattern: 16 DS reads (0x100), then 8 WMMA/MFMA ops (0x8).
    // Keeps the whole B batch live across RA so only the first WMMA of the
    // batch waits on dscnt; the other 7 run with operands already resident.
    SCHED_GROUP(0x100, 16, 0);
    SCHED_GROUP(0x008, 8, 0);
  }

  // D layout: acc[nt].f[i] -> row strip*16 + i + hsel*8, col nt*16 + (lane&15)
#pragma unroll
  for (int nt = 0; nt < 8; ++nt) {
#pragma unroll
    for (int i = 0; i < 8; ++i) {
      long long row = (long long)strip * 16 + i + hsel * 8;
      Y[row * HID + nt * 16 + n0] = acc[nt].f[i];
    }
  }
}

// es[n,h] = sum_c hs[n, h*32+c] * a_s[h*32+c]
__global__ void k_es(const float* __restrict__ hs, const float* __restrict__ as_,
                     float* __restrict__ es, int n) {
  long long idx = (long long)blockIdx.x * blockDim.x + threadIdx.x;
  if (idx >= (long long)n * NHEAD) return;
  int row = (int)(idx >> 2), h = (int)(idx & 3);
  const float* p = hs + (long long)row * HID + h * CH;
  const float* q = as_ + h * CH;
  float acc = 0.f;
#pragma unroll
  for (int c = 0; c < CH; ++c) acc += p[c] * q[c];
  es[idx] = acc;
}

// fold W_dst with a_d:  V[k,h] = sum_c Wd[k, h*32+c] * a_d[h*32+c]
__global__ void k_foldv(const float* __restrict__ Wd, const float* __restrict__ ad_,
                        float* __restrict__ V) {
  int i = blockIdx.x * blockDim.x + threadIdx.x;
  if (i >= HID * NHEAD) return;
  int k = i >> 2, h = i & 3;
  float acc = 0.f;
#pragma unroll
  for (int c = 0; c < CH; ++c) acc += Wd[k * HID + h * CH + c] * ad_[h * CH + c];
  V[k * NHEAD + h] = acc;
}

// ed[n,h] = x[n,:] @ V[:,h]
__global__ void k_ed(const float* __restrict__ x, const float* __restrict__ V,
                     float* __restrict__ ed, int n) {
  long long idx = (long long)blockIdx.x * blockDim.x + threadIdx.x;
  if (idx >= (long long)n * NHEAD) return;
  int row = (int)(idx >> 2), h = (int)(idx & 3);
  const float* p = x + (long long)row * HID;
  float acc = 0.f;
  for (int k = 0; k < HID; ++k) acc += p[k] * V[k * NHEAD + h];
  ed[idx] = acc;
}

__global__ void k_init_ms(uint_t* __restrict__ m, float* __restrict__ s, long long count) {
  long long i = (long long)blockIdx.x * blockDim.x + threadIdx.x;
  if (i < count) { m[i] = 0u; s[i] = 0.0f; }
}

__global__ void k_init_out(float* __restrict__ out, const float* __restrict__ bA,
                           const float* __restrict__ bB, long long count) {
  long long i = (long long)blockIdx.x * blockDim.x + threadIdx.x;
  if (i < count) out[i] = bA[(int)(i & 127)] + bB[(int)(i & 127)];
}

// pass 1: e = leaky_relu(es[src]+ed[dst]); stash e; segment max via atomicMax
__global__ void k_edge1(const int* __restrict__ src, const int* __restrict__ dst,
                        const float* __restrict__ es, const float* __restrict__ ed,
                        float* __restrict__ wbuf, uint_t* __restrict__ m, long long count) {
  long long idx = (long long)blockIdx.x * blockDim.x + threadIdx.x;
  if (idx >= count) return;
  int e = (int)(idx >> 2), h = (int)(idx & 3);
  float v = es[(long long)src[e] * NHEAD + h] + ed[(long long)dst[e] * NHEAD + h];
  v = (v > 0.f) ? v : 0.2f * v;
  wbuf[idx] = v;
  atomicMax(&m[(long long)dst[e] * NHEAD + h], enc_f32(v));
}

// pass 2: w = exp(e - m[dst]); segment sum via atomicAdd
__global__ void k_edge2(const int* __restrict__ dst, const uint_t* __restrict__ m,
                        float* __restrict__ wbuf, float* __restrict__ s, long long count) {
  long long idx = (long long)blockIdx.x * blockDim.x + threadIdx.x;
  if (idx >= count) return;
  int e = (int)(idx >> 2), h = (int)(idx & 3);
  float mm = dec_f32(m[(long long)dst[e] * NHEAD + h]);
  float w = __expf(wbuf[idx] - mm);
  wbuf[idx] = w;
  atomicAdd(&s[(long long)dst[e] * NHEAD + h], w);
}

// pass 2b: alpha = w / (s[dst] + 1e-16)
__global__ void k_edge2b(const int* __restrict__ dst, const float* __restrict__ s,
                         float* __restrict__ wbuf, long long count) {
  long long idx = (long long)blockIdx.x * blockDim.x + threadIdx.x;
  if (idx >= count) return;
  int e = (int)(idx >> 2), h = (int)(idx & 3);
  wbuf[idx] = wbuf[idx] / (s[(long long)dst[e] * NHEAD + h] + 1e-16f);
}

// pass 3: out[dst,:] += hs[src,:] * alpha[edge, head(f)]
__global__ void k_edge3(const int* __restrict__ src, const int* __restrict__ dst,
                        const float* __restrict__ hs, const float* __restrict__ wbuf,
                        float* __restrict__ out, long long count) {
  long long idx = (long long)blockIdx.x * blockDim.x + threadIdx.x;
  if (idx >= count) return;
  int e = (int)(idx >> 7), f = (int)(idx & 127);
  int h = f >> 5;
  float a = wbuf[(long long)e * NHEAD + h];
  float v = hs[(long long)src[e] * HID + f] * a;
  atomicAdd(&out[(long long)dst[e] * HID + f], v);
}

// LayerNorm + exact GELU (+ optional residual). One wave32 per row.
__global__ __launch_bounds__(256) void k_norm(const float* __restrict__ x,
                                              const float* __restrict__ g,
                                              const float* __restrict__ b,
                                              const float* __restrict__ prev,
                                              float* __restrict__ y, int rows) {
  int lane = threadIdx.x & 31;
  int wave = threadIdx.x >> 5;
  int row = blockIdx.x * 8 + wave;
  if (row >= rows) return;
  float v[4];
  float sum = 0.f;
#pragma unroll
  for (int j = 0; j < 4; ++j) {
    v[j] = x[(long long)row * HID + j * 32 + lane];
    sum += v[j];
  }
#pragma unroll
  for (int o = 16; o >= 1; o >>= 1) sum += __shfl_xor(sum, o, 32);
  float mu = sum * (1.0f / 128.0f);
  float sq = 0.f;
#pragma unroll
  for (int j = 0; j < 4; ++j) { float d = v[j] - mu; sq += d * d; }
#pragma unroll
  for (int o = 16; o >= 1; o >>= 1) sq += __shfl_xor(sq, o, 32);
  float inv = rsqrtf(sq * (1.0f / 128.0f) + 1e-5f);
#pragma unroll
  for (int j = 0; j < 4; ++j) {
    int f = j * 32 + lane;
    float t = (v[j] - mu) * inv * g[f] + b[f];
    float ge = 0.5f * t * (1.0f + erff(t * 0.70710678118654752f));
    if (prev) ge += prev[(long long)row * HID + f];
    y[(long long)row * HID + f] = ge;
  }
}

// ---------------- host orchestration ----------------

extern "C" void kernel_launch(void* const* d_in, const int* in_sizes, int n_in,
                              void* d_out, int out_size, void* d_ws, size_t ws_size,
                              hipStream_t stream) {
  (void)n_in; (void)out_size; (void)ws_size;

  const float* x1d = (const float*)d_in[0];
  const float* x2d = (const float*)d_in[1];
  const int* edges[4] = { (const int*)d_in[2], (const int*)d_in[3],
                          (const int*)d_in[4], (const int*)d_in[5] };
  const float* Wp1 = (const float*)d_in[6];
  const float* bp1 = (const float*)d_in[7];
  const float* Wp2 = (const float*)d_in[8];
  const float* bp2 = (const float*)d_in[9];
  const float* W_src   = (const float*)d_in[10];
  const float* W_dst   = (const float*)d_in[11];
  const float* att_src = (const float*)d_in[12];
  const float* att_dst = (const float*)d_in[13];
  const float* bias_g  = (const float*)d_in[14];
  const float* gamma   = (const float*)d_in[15];
  const float* beta    = (const float*)d_in[16];

  const int N1 = in_sizes[0] / 8;
  const int N2 = in_sizes[1] / 6;
  const int E[4] = { in_sizes[2] / 2, in_sizes[3] / 2, in_sizes[4] / 2, in_sizes[5] / 2 };
  int Emax = E[0];
  for (int t = 1; t < 4; ++t) if (E[t] > Emax) Emax = E[t];

  // -------- workspace layout --------
  char* ws = (char*)d_ws;
  size_t off = 0;
  auto carve = [&](size_t bytes) -> void* {
    void* p = ws + off;
    off += (bytes + 255) & ~(size_t)255;
    return p;
  };
  float*    A1  = (float*)carve((size_t)N1 * HID * 4);
  float*    B1  = (float*)carve((size_t)N1 * HID * 4);
  float*    A2  = (float*)carve((size_t)N2 * HID * 4);
  float*    B2  = (float*)carve((size_t)N2 * HID * 4);
  float*    O1  = (float*)carve((size_t)N1 * HID * 4);
  float*    O2  = (float*)carve((size_t)N2 * HID * 4);
  ushort_t* H1B = (ushort_t*)carve((size_t)N1 * HID * 2);
  ushort_t* H2B = (ushort_t*)carve((size_t)N2 * HID * 2);
  float*    HS  = (float*)carve((size_t)N2 * HID * 4);
  float*    ES  = (float*)carve((size_t)N2 * NHEAD * 4);
  float*    ED  = (float*)carve((size_t)N2 * NHEAD * 4);
  uint_t*   M   = (uint_t*)carve((size_t)N2 * NHEAD * 4);
  float*    S   = (float*)carve((size_t)N2 * NHEAD * 4);
  float*    WB  = (float*)carve((size_t)Emax * NHEAD * 4);
  ushort_t* WT  = (ushort_t*)carve((size_t)HID * HID * 2);
  float*    V   = (float*)carve((size_t)HID * NHEAD * 4);

  const int BT = 256;
  const int LDS_BYTES = HID * HID * 2;  // 32 KB weight tile

  // -------- input projections --------
  k_inproj<<<CDIV((long long)N1 * HID, BT), BT, 0, stream>>>(x1d, Wp1, bp1, A1, N1, 8);
  k_inproj<<<CDIV((long long)N2 * HID, BT), BT, 0, stream>>>(x2d, Wp2, bp2, A2, N2, 6);

  float* cur1 = A1; float* alt1 = B1;
  float* cur2 = A2; float* alt2 = B2;

  for (int i = 0; i < 3; ++i) {
    float* y1 = (i == 2) ? (float*)d_out : alt1;
    float* y2 = (i == 2) ? ((float*)d_out + (size_t)N1 * HID) : alt2;

    // bf16-stage current features for the WMMA GEMMs
    k_cvt_bf16<<<CDIV((long long)N1 * HID, BT), BT, 0, stream>>>(cur1, H1B, (long long)N1 * HID);
    k_cvt_bf16<<<CDIV((long long)N2 * HID, BT), BT, 0, stream>>>(cur2, H2B, (long long)N2 * HID);

    // out buffers start at summed biases of their two contributing edge types
    k_init_out<<<CDIV((long long)N1 * HID, BT), BT, 0, stream>>>(
        O1, bias_g + ((size_t)i * 4 + 0) * HID, bias_g + ((size_t)i * 4 + 3) * HID,
        (long long)N1 * HID);
    k_init_out<<<CDIV((long long)N2 * HID, BT), BT, 0, stream>>>(
        O2, bias_g + ((size_t)i * 4 + 1) * HID, bias_g + ((size_t)i * 4 + 2) * HID,
        (long long)N2 * HID);

    auto run_et = [&](int et, const ushort_t* srcb, const float* xdst,
                      int Nsrc, int Ndst, float* out) {
      const float* Ws  = W_src   + ((size_t)i * 4 + et) * HID * HID;
      const float* Wd  = W_dst   + ((size_t)i * 4 + et) * HID * HID;
      const float* as_ = att_src + ((size_t)i * 4 + et) * HID;
      const float* ad_ = att_dst + ((size_t)i * 4 + et) * HID;
      const int* src = edges[et];
      const int* dst = edges[et] + E[et];

      k_wt<<<CDIV(HID * HID, BT), BT, 0, stream>>>(Ws, WT);
      k_gemm_wmma<<<CDIV(Nsrc / 16, 8), BT, LDS_BYTES, stream>>>(srcb, WT, HS, Nsrc);
      k_es<<<CDIV((long long)Nsrc * NHEAD, BT), BT, 0, stream>>>(HS, as_, ES, Nsrc);
      k_foldv<<<CDIV(HID * NHEAD, BT), BT, 0, stream>>>(Wd, ad_, V);
      k_ed<<<CDIV((long long)Ndst * NHEAD, BT), BT, 0, stream>>>(xdst, V, ED, Ndst);
      k_init_ms<<<CDIV((long long)Ndst * NHEAD, BT), BT, 0, stream>>>(M, S, (long long)Ndst * NHEAD);

      long long eh = (long long)E[et] * NHEAD;
      k_edge1<<<CDIV(eh, BT), BT, 0, stream>>>(src, dst, ES, ED, WB, M, eh);
      k_edge2<<<CDIV(eh, BT), BT, 0, stream>>>(dst, M, WB, S, eh);
      k_edge2b<<<CDIV(eh, BT), BT, 0, stream>>>(dst, S, WB, eh);
      long long ef = (long long)E[et] * HID;
      k_edge3<<<CDIV(ef, BT), BT, 0, stream>>>(src, dst, HS, WB, out, ef);
    };

    // '1d' receives pipe(0: h1->h1) + couples_from(3: h2->h1)
    run_et(0, H1B, cur1, N1, N1, O1);
    run_et(3, H2B, cur1, N2, N1, O1);
    // '2d' receives surface(1: h2->h2) + couples_to(2: h1->h2)
    run_et(1, H2B, cur2, N2, N2, O2);
    run_et(2, H1B, cur2, N1, N2, O2);

    const float* prev1 = (i < 2) ? cur1 : nullptr;
    const float* prev2 = (i < 2) ? cur2 : nullptr;
    k_norm<<<CDIV(N1, 8), BT, 0, stream>>>(O1, gamma + ((size_t)i * 2 + 0) * HID,
                                           beta + ((size_t)i * 2 + 0) * HID, prev1, y1, N1);
    k_norm<<<CDIV(N2, 8), BT, 0, stream>>>(O2, gamma + ((size_t)i * 2 + 1) * HID,
                                           beta + ((size_t)i * 2 + 1) * HID, prev2, y2, N2);

    if (i < 2) {
      float* t1 = cur1; cur1 = alt1; alt1 = t1;
      float* t2 = cur2; cur2 = alt2; alt2 = t2;
    }
  }
}
